// Self_Attention_47304769798930
// MI455X (gfx1250) — compile-verified, matrix operand order checked by reading, not run on hardware
//
#include <hip/hip_runtime.h>
#include <hip/hip_bf16.h>

// Shapes from the reference
#define NCI 64      // input channels
#define CB  8       // bottleneck channels
#define NB  4       // batch
#define NN  4096    // 16*16*16 voxels
#define TS  16      // WMMA tile edge

// LDS layout for the attention kernel (dynamic LDS only => offsets start at 0)
#define GS_OFF   0            // g  staging: [NN][8] f16  = 65536 B
#define HS_OFF   65536        // h  staging: [NN][8] f16  = 65536 B
#define PS_OFF   131072       // p  staging: 4 waves * 16*32 f16 = 4096 B
#define ZP_OFF   135168       // zero pad region: 512 B
#define SMEM_SZ  135680

typedef _Float16 v16h __attribute__((ext_vector_type(16)));
typedef _Float16 v8h  __attribute__((ext_vector_type(8)));
typedef float    v8f  __attribute__((ext_vector_type(8)));
typedef unsigned int v4u __attribute__((ext_vector_type(4)));
typedef int      v8i  __attribute__((ext_vector_type(8)));
typedef int      v4i  __attribute__((ext_vector_type(4)));

__device__ __forceinline__ v8f zero8f() {
    v8f z;
#pragma unroll
    for (int i = 0; i < 8; ++i) z[i] = 0.0f;
    return z;
}

// ---------------------------------------------------------------------------
// Kernel 1: per-voxel projections  f,g,h = W @ x + b  -> f16, layout [b][n][8]
// ---------------------------------------------------------------------------
__global__ void proj_fgh(const float* __restrict__ x,
                         const float* __restrict__ Wf, const float* __restrict__ bfp,
                         const float* __restrict__ Wg, const float* __restrict__ bgp,
                         const float* __restrict__ Wh, const float* __restrict__ bhp,
                         uint4* __restrict__ fT, uint4* __restrict__ gT,
                         uint4* __restrict__ hT)
{
    __shared__ float sWf[CB * NCI], sWg[CB * NCI], sWh[CB * NCI];
    __shared__ float sbf[CB], sbg[CB], sbh[CB];
    for (int i = threadIdx.x; i < CB * NCI; i += blockDim.x) {
        sWf[i] = Wf[i]; sWg[i] = Wg[i]; sWh[i] = Wh[i];
    }
    if (threadIdx.x < CB) {
        sbf[threadIdx.x] = bfp[threadIdx.x];
        sbg[threadIdx.x] = bgp[threadIdx.x];
        sbh[threadIdx.x] = bhp[threadIdx.x];
    }
    __syncthreads();

    const int idx = blockIdx.x * blockDim.x + threadIdx.x;   // 0 .. B*N-1
    const int b = idx >> 12;
    const int n = idx & (NN - 1);

    float af[CB], ag[CB], ah[CB];
#pragma unroll
    for (int o = 0; o < CB; ++o) { af[o] = sbf[o]; ag[o] = sbg[o]; ah[o] = sbh[o]; }

    const float* xp = x + (size_t)b * NCI * NN + n;
#pragma unroll 8
    for (int c = 0; c < NCI; ++c) {
        const float xv = xp[(size_t)c * NN];
#pragma unroll
        for (int o = 0; o < CB; ++o) {
            af[o] = fmaf(sWf[o * NCI + c], xv, af[o]);
            ag[o] = fmaf(sWg[o * NCI + c], xv, ag[o]);
            ah[o] = fmaf(sWh[o * NCI + c], xv, ah[o]);
        }
    }

    union { uint4 u; _Float16 h[8]; } pf, pg, ph;
#pragma unroll
    for (int o = 0; o < CB; ++o) {
        pf.h[o] = (_Float16)af[o];
        pg.h[o] = (_Float16)ag[o];
        ph.h[o] = (_Float16)ah[o];
    }
    fT[(size_t)b * NN + n] = pf.u;
    gT[(size_t)b * NN + n] = pg.u;
    hT[(size_t)b * NN + n] = ph.u;
}

// ---------------------------------------------------------------------------
// TDM: one descriptor moves a contiguous 64KB span (8192 x 8B) into LDS.
// Kernel uses only dynamic LDS, so LDS byte offsets are absolute.
// Toolchain here exposes the 6-arg form:
//   (uint32x4 g0, int32x8 g1, int32x4 g2, int32x4 g3, int32x8 pad, i32 cpol)
// ---------------------------------------------------------------------------
#if __has_builtin(__builtin_amdgcn_tensor_load_to_lds)
#define HAVE_TDM 1
__device__ __forceinline__ void tdm_load_64k(unsigned lds_off, const void* gsrc) {
    const unsigned long long ga = (unsigned long long)gsrc;
    v4u g0;
    g0[0] = 1u;                                       // count=1, no gather
    g0[1] = lds_off;                                  // lds_addr
    g0[2] = (unsigned)(ga & 0xffffffffu);             // global_addr[31:0]
    g0[3] = (unsigned)((ga >> 32) & 0x01ffffffu)      // global_addr[56:32]
          | (2u << 30);                               // type = 2 (image)
    v8i g1;
    g1[0] = (int)(3u << 16);                          // data_size = 8B
    g1[1] = (int)(0x2000u << 16);                     // tensor_dim0 = 8192 (lo16)
    g1[2] = (int)(0x0001u << 16);                     // dim0 hi16=0 ; tensor_dim1 = 1
    g1[3] = (int)(0x2000u << 16);                     // dim1 hi16=0 ; tile_dim0 = 8192
    g1[4] = 1;                                        // tile_dim1 = 1 ; tile_dim2 = 0
    g1[5] = 0x2000;                                   // tensor_dim0_stride = 8192
    g1[6] = 0;
    g1[7] = 0;
    const v4i z4 = {0, 0, 0, 0};
    const v8i z8 = {0, 0, 0, 0, 0, 0, 0, 0};
    __builtin_amdgcn_tensor_load_to_lds(g0, g1, z4, z4, z8, 0);
}
#else
#define HAVE_TDM 0
#endif

// ---------------------------------------------------------------------------
// Kernel 2: fused attention.  One wave owns 16 softmax rows.
//   s = f^T g (WMMA f16, K=8 zero-padded to 32) -> softmax -> o = p h^T (WMMA)
// ---------------------------------------------------------------------------
__global__ void attn_fused(const uint4* __restrict__ fT, const uint4* __restrict__ gT,
                           const uint4* __restrict__ hT, float* __restrict__ oT)
{
    extern __shared__ char smem[];
    _Float16* gS   = (_Float16*)(smem + GS_OFF);   // [NN][8] f16
    _Float16* hS   = (_Float16*)(smem + HS_OFF);   // [NN][8] f16
    _Float16* pS   = (_Float16*)(smem + PS_OFF);   // [4][16][32] f16
    _Float16* zpad = (_Float16*)(smem + ZP_OFF);   // 512 B of zeros

    const int tid  = threadIdx.x;
    const int wave = tid >> 5;
    const int lane = tid & 31;
    const int b    = blockIdx.x >> 6;              // 4 batches * 64 groups
    const int tg   = blockIdx.x & 63;
    const int n0   = (tg * 4 + wave) * TS;         // this wave's 16 rows

    // zero-pad region (feeds the zero halves of WMMA operands via addressing)
    if (tid < 128) ((float*)zpad)[tid] = 0.0f;

    // ---- stage g,h for this batch into LDS
#if HAVE_TDM
    if (wave == 0) {
        tdm_load_64k(GS_OFF, gT + (size_t)b * NN);
        tdm_load_64k(HS_OFF, hT + (size_t)b * NN);
        __builtin_amdgcn_s_wait_tensorcnt(0);
    }
#else
    {
        uint4* gS4 = (uint4*)gS;
        uint4* hS4 = (uint4*)hS;
        for (int i = tid; i < NN; i += blockDim.x) {
            gS4[i] = gT[(size_t)b * NN + i];
            hS4[i] = hT[(size_t)b * NN + i];
        }
    }
#endif
    __syncthreads();

    _Float16* pW = pS + wave * (TS * 32);

    const int cc    = lane & 15;                   // column within tile / B-col
    const int kb    = (lane < 16) ? 0 : 8;         // K sub-range per lane half
    const int rbase = (lane >> 4) << 3;            // C-layout row offset (0 or 8)

    // ---- A operand: f^T tile, rows n0..n0+15, K = 8 channels padded to 32
    v16h af;
    {
        union { uint4 u; _Float16 h[8]; } t;
        t.u = fT[(size_t)b * NN + n0 + cc];
#pragma unroll
        for (int k = 0; k < 8; ++k) {
            af[k]     = (lane < 16) ? t.h[k] : (_Float16)0.0f;
            af[8 + k] = (_Float16)0.0f;
        }
    }

    // per-lane B-operand streams: invalid lane halves walk the LDS zero region
    const _Float16* gp   = (lane < 16) ? (gS + (size_t)cc * CB) : zpad;
    const int       gstp = (lane < 16) ? (TS * CB) : 0;

    // ---- Pass 1: row maxima over all 4096 columns
    float mx[8];
#pragma unroll
    for (int r = 0; r < 8; ++r) mx[r] = -3.0e38f;

    {
        const _Float16* g1p = gp;
        for (int mt = 0; mt < NN / TS; ++mt) {
            const v8h gv = *(const v8h*)g1p;
            g1p += gstp;
            v16h bg;
#pragma unroll
            for (int k = 0; k < 8; ++k) { bg[k] = gv[k]; bg[8 + k] = (_Float16)0.0f; }
            v8f s = __builtin_amdgcn_wmma_f32_16x16x32_f16(
                false, af, false, bg, (short)0, zero8f(), false, false);
#pragma unroll
            for (int r = 0; r < 8; ++r) mx[r] = fmaxf(mx[r], s[r]);
        }
    }
#pragma unroll
    for (int r = 0; r < 8; ++r)
        for (int off = 8; off; off >>= 1)
            mx[r] = fmaxf(mx[r], __shfl_xor(mx[r], off, 32));

    // ---- Pass 2: p = exp(s - max); accumulate row sums and o = p @ h^T
    float sm[8];
#pragma unroll
    for (int r = 0; r < 8; ++r) sm[r] = 0.0f;
    v8f oacc = zero8f();

    const bool       cv   = cc < CB;
    const _Float16*  hp   = cv ? (hS + cc) : zpad;  // gathers up to 496B into zpad
    const int        hstp = cv ? (32 * CB) : 0;
    const _Float16*  g2p  = gp;

    for (int mp = 0; mp < NN / 32; ++mp) {
#pragma unroll
        for (int t = 0; t < 2; ++t) {
            const v8h gv = *(const v8h*)(g2p + t * gstp);
            v16h bg;
#pragma unroll
            for (int k = 0; k < 8; ++k) { bg[k] = gv[k]; bg[8 + k] = (_Float16)0.0f; }
            v8f s = __builtin_amdgcn_wmma_f32_16x16x32_f16(
                false, af, false, bg, (short)0, zero8f(), false, false);
#pragma unroll
            for (int r = 0; r < 8; ++r) {
                const float pv = __expf(s[r] - mx[r]);
                sm[r] += pv;
                pW[(r + rbase) * 32 + cc + t * TS] = (_Float16)pv;  // C-layout -> row-major
            }
        }
        g2p += 2 * gstp;

        // A operand: p tile, 16 rows x K=32, from row-major staging
        const v8h plo = *(const v8h*)(pW + cc * 32 + kb);
        const v8h phi = *(const v8h*)(pW + cc * 32 + 16 + kb);
        v16h ap;
#pragma unroll
        for (int k = 0; k < 8; ++k) { ap[k] = plo[k]; ap[8 + k] = phi[k]; }

        // B operand: h^T, K=32 rows (m), N=16 cols (c: 8 real + 8 zero via zpad)
        v16h bh;
#pragma unroll
        for (int k = 0; k < 8; ++k) {
            bh[k]     = hp[(kb + k) * CB];
            bh[8 + k] = hp[(16 + kb + k) * CB];
        }
        hp += hstp;

        oacc = __builtin_amdgcn_wmma_f32_16x16x32_f16(
            false, ap, false, bh, (short)0, oacc, false, false);
    }

#pragma unroll
    for (int r = 0; r < 8; ++r)
        for (int off = 8; off; off >>= 1)
            sm[r] += __shfl_xor(sm[r], off, 32);

    // write o[b][n][c] normalized by the row sum
    if (cv) {
#pragma unroll
        for (int r = 0; r < 8; ++r) {
            const int row = r + rbase;
            oT[((size_t)b * NN + n0 + row) * CB + cc] = oacc[r] / sm[r];
        }
    }
}

// ---------------------------------------------------------------------------
// Kernel 3: out = gamma * (Wv @ o + bv) + x
// ---------------------------------------------------------------------------
__global__ void out_proj(const float* __restrict__ oT, const float* __restrict__ Wv,
                         const float* __restrict__ bv, const float* __restrict__ gammap,
                         const float* __restrict__ x, float* __restrict__ out)
{
    __shared__ float sW[NCI * CB];
    __shared__ float sb[NCI];
    for (int i = threadIdx.x; i < NCI * CB; i += blockDim.x) sW[i] = Wv[i];
    if (threadIdx.x < NCI) sb[threadIdx.x] = bv[threadIdx.x];
    __syncthreads();

    const int idx = blockIdx.x * blockDim.x + threadIdx.x;
    const int b = idx >> 12;
    const int n = idx & (NN - 1);
    const float g = gammap[0];

    float ov[CB];
    const float* op = oT + ((size_t)b * NN + n) * CB;
#pragma unroll
    for (int c = 0; c < CB; ++c) ov[c] = op[c];

#pragma unroll 4
    for (int co = 0; co < NCI; ++co) {
        float a = sb[co];
#pragma unroll
        for (int c = 0; c < CB; ++c) a = fmaf(sW[co * CB + c], ov[c], a);
        const size_t xi = ((size_t)b * NCI + co) * NN + n;
        out[xi] = g * a + x[xi];
    }
}

// ---------------------------------------------------------------------------
extern "C" void kernel_launch(void* const* d_in, const int* in_sizes, int n_in,
                              void* d_out, int out_size, void* d_ws, size_t ws_size,
                              hipStream_t stream) {
    const float* x     = (const float*)d_in[0];
    const float* Wf    = (const float*)d_in[1];
    const float* bf    = (const float*)d_in[2];
    const float* Wg    = (const float*)d_in[3];
    const float* bg    = (const float*)d_in[4];
    const float* Wh    = (const float*)d_in[5];
    const float* bh    = (const float*)d_in[6];
    const float* Wv    = (const float*)d_in[7];
    const float* bv    = (const float*)d_in[8];
    const float* gamma = (const float*)d_in[9];

    char* ws = (char*)d_ws;
    uint4* fT = (uint4*)(ws);                       // B*N rows * 16B  = 256 KB
    uint4* gT = (uint4*)(ws + 262144);              // 256 KB
    uint4* hT = (uint4*)(ws + 524288);              // 256 KB
    float* oT = (float*)(ws + 786432);              // B*N*8 f32 = 512 KB

    proj_fgh<<<(NB * NN) / 256, 256, 0, stream>>>(x, Wf, bf, Wg, bg, Wh, bh,
                                                  fT, gT, hT);

    attn_fused<<<NB * (NN / TS / 4), 128, SMEM_SZ, stream>>>(fT, gT, hT, oT);

    out_proj<<<(NB * NN) / 256, 256, 0, stream>>>(oT, Wv, bv, gamma, x,
                                                  (float*)d_out);
}
